// GNNModel_70437463654489
// MI455X (gfx1250) — matrix operand order repaired and checked by reading, not compiled, not run on hardware
//
#include <hip/hip_runtime.h>

#define NNODES 100000
#define FDIM   256
#define HDIM   256
#define CDIM   64
#define MTILES (NNODES / 16)

typedef __attribute__((ext_vector_type(16))) _Float16 v16h;
typedef __attribute__((ext_vector_type(8)))  float    v8f;

// ---------------- degree / norm ----------------
__global__ void init_deg_kernel(float* deg, int n) {
    int i = blockIdx.x * blockDim.x + threadIdx.x;
    if (i < n) deg[i] = 1.0f;                       // self-loop contribution
}

__global__ void count_deg_kernel(const long long* __restrict__ dst, float* deg, int e) {
    int i = blockIdx.x * blockDim.x + threadIdx.x;
    if (i < e) atomicAdd(&deg[(int)dst[i]], 1.0f);
}

__global__ void make_dis_kernel(const float* __restrict__ deg, float* dis, int n) {
    int i = blockIdx.x * blockDim.x + threadIdx.x;
    if (i < n) dis[i] = rsqrtf(deg[i]);             // deg >= 1 always
}

// ---------------- LDS-staged, register-blocked WMMA GEMM ----------------
// C[M, NCOL] = A[M, 256] @ B[256, NCOL] (+bias)(+relu),  NCOL = TILES_N*16.
// Block = 8 waves covering (8/WPM) M-tiles x WPM N-slices, WPM = TILES_N/TPW.
// Each wave owns a 16 x (TPW*16) strip: TPW accumulators = TPW*8 VGPRs
// (TPW=8 -> 64 acc VGPRs: no scratch spills, unlike full-row blocking).
// K staged in 64-row chunks: block converts B slice to f16 in LDS, pre-swizzled
// into WMMA B-fragment order [s][tn][lane][16 f16], so the hot loop reads each
// fragment as two contiguous ds_load_b128 and reuses one A fragment for TPW WMMAs.
// Fragment layouts per CDNA5 ISA 7.12.2 (wave32):
//   A (16x32 f16): lanes 0-15 row=lane, elems 0..7=K0..7, 8..15=K16..23;
//                  lanes 16-31 same rows, K8..15 / K24..31.
//   B (32x16 f16): lane&15 = column; same K split as A (lanes 0-15 K0..7/K16..23, lanes 16-31 K8..15/K24..31).
//   C/D (16x16 f32): VGPR v -> row v (+8 for lanes>=16), lane&15 = column.
template <int TILES_N, int TPW>
__global__ void gemm_wmma_lds(const float* __restrict__ A, const float* __restrict__ B,
                              const float* __restrict__ bias, float* __restrict__ C,
                              int Mtiles, int addBias, int doRelu) {
    constexpr int NCOL  = TILES_N * 16;
    constexpr int WPM   = TILES_N / TPW;                    // waves per M-tile
    constexpr int MPB   = 8 / WPM;                          // M-tiles per block
    constexpr int SLOTS = 2 * TILES_N * 32;                 // lane-slots per 64-row chunk
    constexpr int SPT   = (SLOTS + 255) / 256;              // slots per thread

    __shared__ __align__(32) _Float16 Bs[SLOTS * 16];       // 32 KB (TN=16) / 8 KB (TN=4)

    const int  tid    = threadIdx.x;
    const int  lane   = tid & 31;
    const int  wid    = tid >> 5;
    const int  tm     = blockIdx.x * MPB + wid / WPM;
    const int  tnBase = (wid % WPM) * TPW;
    const bool active = (tm < Mtiles);

    const int half = (lane < 16) ? 0 : 8;                   // A K-half / D row-half select
    const float* Arow = A + (size_t)(active ? (tm * 16 + (lane & 15)) : 0) * 256;

    v8f zero = {};
    v8f acc[TPW];
#pragma unroll
    for (int t = 0; t < TPW; ++t) acc[t] = zero;

    for (int kc = 0; kc < 256; kc += 64) {
        // ---- stage: B[kc..kc+63, :] -> LDS in fragment order ----
#pragma unroll
        for (int i = 0; i < SPT; ++i) {
            int slot = tid + i * 256;
            if (slot < SLOTS) {
                int L  = slot & 31;
                int tn = (slot >> 5) % TILES_N;
                int s  = slot / (32 * TILES_N);
                int n  = tn * 16 + (L & 15);
                int kb = kc + s * 32 + ((L < 16) ? 0 : 8);
                _Float16* dstp = &Bs[(size_t)slot * 16];
#pragma unroll
                for (int j = 0; j < 16; ++j)
                    dstp[j] = (_Float16)B[(size_t)(kb + j + ((j >= 8) ? 8 : 0)) * NCOL + n];
            }
        }
        __syncthreads();

        if (active) {
#pragma unroll
            for (int s = 0; s < 2; ++s) {
                const int k0 = kc + s * 32;
                const float4* ap0 = reinterpret_cast<const float4*>(Arow + k0 + half);
                const float4* ap1 = reinterpret_cast<const float4*>(Arow + k0 + 16 + half);
                float4 a0 = ap0[0], a1 = ap0[1];
                float4 a2 = ap1[0], a3 = ap1[1];
                float at[16] = {a0.x,a0.y,a0.z,a0.w, a1.x,a1.y,a1.z,a1.w,
                                a2.x,a2.y,a2.z,a2.w, a3.x,a3.y,a3.z,a3.w};
                v16h af;
#pragma unroll
                for (int j = 0; j < 16; ++j) af[j] = (_Float16)at[j];
#pragma unroll
                for (int ti = 0; ti < TPW; ++ti) {
                    const int tn = tnBase + ti;
                    const v16h* bp = reinterpret_cast<const v16h*>(
                        &Bs[(size_t)((s * TILES_N + tn) * 32 + lane) * 16]);
                    v16h bf = *bp;                          // 2x ds_load_b128
                    acc[ti] = __builtin_amdgcn_wmma_f32_16x16x32_f16(
                                  false, af, false, bf, (short)0, acc[ti], false, false);
                }
            }
        }
        __syncthreads();
    }

    if (!active) return;
    const int rbase = tm * 16 + half;
#pragma unroll
    for (int ti = 0; ti < TPW; ++ti) {
        const int col = (tnBase + ti) * 16 + (lane & 15);
        const float bv = addBias ? bias[col] : 0.0f;
#pragma unroll
        for (int v = 0; v < 8; ++v) {
            float val = acc[ti][v] + bv;
            if (doRelu) val = fmaxf(val, 0.0f);
            C[(size_t)(rbase + v) * NCOL + col] = val;
        }
    }
}

// ---------------- aggregation ----------------
// agg[i,:] = t[i,:] * dis[i]^2   (self-loop term; initializes accumulator)
__global__ void agg_init_kernel(const float* __restrict__ t, const float* __restrict__ dis,
                                float* __restrict__ agg) {
    int row = blockIdx.x, f = threadIdx.x;
    float di = dis[row];
    size_t idx = (size_t)row * HDIM + f;
    agg[idx] = t[idx] * di * di;
}

// One wave32 per edge: agg[dst,:] += t[src,:] * dis[src]*dis[dst]
// 32 lanes x 8 floats; float4 gathers + f32 atomics (t and agg are L2-resident).
__global__ void agg_edges_kernel(const long long* __restrict__ src,
                                 const long long* __restrict__ dst,
                                 const float* __restrict__ dis,
                                 const float* __restrict__ t,
                                 float* __restrict__ agg, int e) {
    int gid  = blockIdx.x * blockDim.x + threadIdx.x;
    int edge = gid >> 5;
    int lane = gid & 31;
    if (edge >= e) return;
    int s = (int)src[edge];
    int d = (int)dst[edge];
    float w = dis[s] * dis[d];
    const float4* tp = reinterpret_cast<const float4*>(t + (size_t)s * HDIM + lane * 8);
    float4 v0 = tp[0], v1 = tp[1];
    float* ap = agg + (size_t)d * HDIM + lane * 8;
    atomicAdd(ap + 0, v0.x * w); atomicAdd(ap + 1, v0.y * w);
    atomicAdd(ap + 2, v0.z * w); atomicAdd(ap + 3, v0.w * w);
    atomicAdd(ap + 4, v1.x * w); atomicAdd(ap + 5, v1.y * w);
    atomicAdd(ap + 6, v1.z * w); atomicAdd(ap + 7, v1.w * w);
}

// h = relu(agg + b), in place
__global__ void bias_relu_kernel(float* __restrict__ h, const float* __restrict__ b) {
    int row = blockIdx.x, f = threadIdx.x;
    size_t idx = (size_t)row * HDIM + f;
    h[idx] = fmaxf(h[idx] + b[f], 0.0f);
}

// ---------------- driver ----------------
extern "C" void kernel_launch(void* const* d_in, const int* in_sizes, int n_in,
                              void* d_out, int out_size, void* d_ws, size_t ws_size,
                              hipStream_t stream) {
    const float*     x   = (const float*)d_in[0];
    const long long* ei  = (const long long*)d_in[1];   // int64 edge_index [2,E]
    const float*     W1  = (const float*)d_in[2];
    const float*     b1  = (const float*)d_in[3];
    const float*     W2  = (const float*)d_in[4];
    const float*     b2  = (const float*)d_in[5];
    const float*     W3  = (const float*)d_in[6];
    const float*     b3  = (const float*)d_in[7];
    const float*     Wl  = (const float*)d_in[8];
    const float*     bl  = (const float*)d_in[9];
    float*           out = (float*)d_out;

    const int E = in_sizes[1] / 2;
    const long long* srcp = ei;
    const long long* dstp = ei + E;

    char*  ws = (char*)d_ws;
    const size_t NH = (size_t)NNODES * HDIM * sizeof(float);   // 102,400,000 B
    float* t   = (float*)(ws);
    float* agg = (float*)(ws + NH);
    float* deg = (float*)(ws + 2 * NH);
    float* dis = deg + NNODES;

    // norm
    init_deg_kernel<<<(NNODES + 255) / 256, 256, 0, stream>>>(deg, NNODES);
    count_deg_kernel<<<(E + 255) / 256, 256, 0, stream>>>(dstp, deg, E);
    make_dis_kernel<<<(NNODES + 255) / 256, 256, 0, stream>>>(deg, dis, NNODES);

    const int gemmBlksH = (MTILES + 3) / 4;                     // <16,8>: 4 M-tiles/block
    const int gemmBlksL = (MTILES + 7) / 8;                     // <4,4>:  8 M-tiles/block
    const int edgeBlks  = (int)(((size_t)E * 32 + 255) / 256);

    // layer 1
    gemm_wmma_lds<16, 8><<<gemmBlksH, 256, 0, stream>>>(x,   W1, nullptr, t, MTILES, 0, 0);
    agg_init_kernel<<<NNODES, 256, 0, stream>>>(t, dis, agg);
    agg_edges_kernel<<<edgeBlks, 256, 0, stream>>>(srcp, dstp, dis, t, agg, E);
    bias_relu_kernel<<<NNODES, 256, 0, stream>>>(agg, b1);
    // layer 2
    gemm_wmma_lds<16, 8><<<gemmBlksH, 256, 0, stream>>>(agg, W2, nullptr, t, MTILES, 0, 0);
    agg_init_kernel<<<NNODES, 256, 0, stream>>>(t, dis, agg);
    agg_edges_kernel<<<edgeBlks, 256, 0, stream>>>(srcp, dstp, dis, t, agg, E);
    bias_relu_kernel<<<NNODES, 256, 0, stream>>>(agg, b2);
    // layer 3
    gemm_wmma_lds<16, 8><<<gemmBlksH, 256, 0, stream>>>(agg, W3, nullptr, t, MTILES, 0, 0);
    agg_init_kernel<<<NNODES, 256, 0, stream>>>(t, dis, agg);
    agg_edges_kernel<<<edgeBlks, 256, 0, stream>>>(srcp, dstp, dis, t, agg, E);
    bias_relu_kernel<<<NNODES, 256, 0, stream>>>(agg, b3);
    // classifier: out = h3 @ Wl + bl   [N, 64]
    gemm_wmma_lds<4, 4><<<gemmBlksL, 256, 0, stream>>>(agg, Wl, bl, out, MTILES, 1, 0);
}
